// LongConv_35459249995892
// MI455X (gfx1250) — compile-verified, matrix operand order checked by reading, not run on hardware
//
#include <hip/hip_runtime.h>

// Problem constants (from reference): B=2, L=4096, D=1024, C=4
#define Bz 2
#define Lz 4096
#define Dz 1024
#define Cz 4
#define CDz (Cz * Dz)
#define Mz (Bz * Lz)

typedef __attribute__((ext_vector_type(16))) __bf16 v16bf;
typedef __attribute__((ext_vector_type(8)))  float  v8f;
typedef __attribute__((ext_vector_type(4)))  unsigned int v4u;
typedef __attribute__((ext_vector_type(8)))  int v8i_t;
typedef __attribute__((ext_vector_type(4)))  int v4i_t;

union Frag16 {
    v16bf v;
    unsigned short h[16];
    unsigned int   w[8];
};

__device__ __forceinline__ unsigned short f2bf(float f) {
    unsigned int x = __float_as_uint(f);
    x += 0x7FFFu + ((x >> 16) & 1u);   // round-to-nearest-even
    return (unsigned short)(x >> 16);
}

// ---------------------------------------------------------------------------
// Tensor Data Mover helpers (gfx1250). Guarded so the build never regresses.
// This toolchain exposes the 6-arg builtin:
//   (uint32x4 g0, int32x8 g1, int32x4 g2, int32x4 g3, int32x8 g4, i32 cpol)
// ---------------------------------------------------------------------------
#if __has_builtin(__builtin_amdgcn_tensor_load_to_lds) && \
    __has_builtin(__builtin_amdgcn_s_wait_tensorcnt)
#define USE_TDM_LOAD 1
#else
#define USE_TDM_LOAD 0
#endif

#if __has_builtin(__builtin_amdgcn_tensor_store_from_lds) && \
    __has_builtin(__builtin_amdgcn_s_wait_tensorcnt)
#define USE_TDM_STORE 1
#else
#define USE_TDM_STORE 0
#endif

#if USE_TDM_LOAD || USE_TDM_STORE
__device__ __forceinline__ unsigned lds_offset_of(const void* p) {
    // generic -> LDS addrspace cast yields the 32-bit LDS byte offset
    return (unsigned)(unsigned long long)
        (const __attribute__((address_space(3))) void*)p;
}

// Build D# groups for a contiguous 1-D row transfer of `nelem` elements.
// data_size_code: 0=1B, 1=2B, 2=4B (ISA 8.4).
__device__ __forceinline__ void tdm_make_desc(
    const void* gaddr, unsigned lds_off, unsigned nelem,
    unsigned data_size_code, v4u* g0, v8i_t* g1)
{
    unsigned long long ga = (unsigned long long)gaddr;
    v4u a;
    a[0] = 1u;                                    // count=1, user descriptor
    a[1] = lds_off;                               // lds_addr (bytes)
    a[2] = (unsigned)(ga & 0xFFFFFFFFu);          // global_addr[31:0]
    a[3] = (unsigned)((ga >> 32) & 0x01FFFFFFu)   // global_addr[56:32]
         | (2u << 30);                            // type = 2 ("image")
    v8i_t b;
    b[0] = (int)(data_size_code << 16);           // wg_mask=0, data_size
    b[1] = (int)((nelem & 0xFFFFu) << 16);        // tensor_dim0[15:0]
    b[2] = (int)(((nelem >> 16) & 0xFFFFu)        // tensor_dim0[31:16]
         | (1u << 16));                           // tensor_dim1 = 1
    b[3] = (int)((nelem & 0xFFFFu) << 16);        // tile_dim0 = nelem
    b[4] = 0;                                     // tile_dim1/2 unused (1-D)
    b[5] = (int)nelem;                            // tensor_dim0_stride[31:0]
    b[6] = 0;                                     // stride0 hi | stride1 lo
    b[7] = 0;
    *g0 = a; *g1 = b;
}
#endif

// ---------------------------------------------------------------------------
// Kernel 1: in_proj  x = u @ in_w + in_b, written TRANSPOSED as xt[B, D, L] f32
// bf16 WMMA, 128x128 tile / workgroup, 8 waves (wave32), 2x4 accum tiles/wave.
// ---------------------------------------------------------------------------
__global__ __launch_bounds__(256) void in_proj_wmma(
    const float* __restrict__ u, const float* __restrict__ w,
    const float* __restrict__ bias, float* __restrict__ xt)
{
    __shared__ unsigned short sA[128][34];   // [m][k] bf16
    __shared__ unsigned short sB[128][34];   // [n][k] bf16 (pre-transposed)
    __shared__ float          sT[128][33];   // transpose staging (32-col chunks)

    const int tid    = threadIdx.x;
    const int lane   = tid & 31;
    const int wid    = tid >> 5;
    const int wm     = wid & 3;      // 4 row-groups of 32
    const int wn     = wid >> 2;     // 2 col-groups of 64
    const int m0     = blockIdx.y * 128;
    const int n0     = blockIdx.x * 128;
    const int lane16 = lane & 15;
    const int koff   = (lane < 16) ? 0 : 8;

    v8f acc[2][4];
    const v8f vzero = {0.f, 0.f, 0.f, 0.f, 0.f, 0.f, 0.f, 0.f};
    for (int mi = 0; mi < 2; ++mi)
        for (int ni = 0; ni < 4; ++ni) acc[mi][ni] = vzero;

    for (int kt = 0; kt < Dz / 32; ++kt) {
        __syncthreads();
        {   // stage A tile 128x32 from u[M, D]
            int flat = tid * 16;
            int ml = flat >> 5, kk = flat & 31;
            const float* src = &u[(size_t)(m0 + ml) * Dz + kt * 32 + kk];
            #pragma unroll
            for (int i = 0; i < 16; ++i) sA[ml][kk + i] = f2bf(src[i]);
        }
        {   // stage B tile 32x128 from in_w[D, D] -> sB[n][k]
            int flat = tid * 16;
            int kk = flat >> 7, nl = flat & 127;
            const float* src = &w[(size_t)(kt * 32 + kk) * Dz + n0 + nl];
            #pragma unroll
            for (int i = 0; i < 16; ++i) sB[nl + i][kk] = f2bf(src[i]);
        }
        if (kt + 1 < Dz / 32) {   // global_prefetch_b8 of next tiles
            __builtin_prefetch(&u[(size_t)(m0 + (tid >> 1)) * Dz + (kt + 1) * 32], 0, 0);
            __builtin_prefetch(&w[(size_t)((kt + 1) * 32 + (tid >> 4)) * Dz + n0], 0, 0);
        }
        __syncthreads();

        Frag16 a[2], b[4];
        #pragma unroll
        for (int mi = 0; mi < 2; ++mi) {
            int row = wm * 32 + mi * 16 + lane16;
            #pragma unroll
            for (int j = 0; j < 4; ++j) {
                a[mi].w[j]     = *(const unsigned int*)&sA[row][koff + j * 2];
                a[mi].w[j + 4] = *(const unsigned int*)&sA[row][koff + 16 + j * 2];
            }
        }
        #pragma unroll
        for (int ni = 0; ni < 4; ++ni) {
            int col = wn * 64 + ni * 16 + lane16;
            #pragma unroll
            for (int j = 0; j < 4; ++j) {
                b[ni].w[j]     = *(const unsigned int*)&sB[col][koff + j * 2];
                b[ni].w[j + 4] = *(const unsigned int*)&sB[col][koff + 16 + j * 2];
            }
        }
        #pragma unroll
        for (int mi = 0; mi < 2; ++mi)
            #pragma unroll
            for (int ni = 0; ni < 4; ++ni)
                acc[mi][ni] = __builtin_amdgcn_wmma_f32_16x16x32_bf16(
                    false, a[mi].v, false, b[ni].v, (short)0, acc[mi][ni],
                    false, false);
    }

    // epilogue: bias add + LDS transpose, store xt[B, D, L] coalesced along L
    const int bidx  = m0 >> 12;        // /L
    const int lbase = m0 & (Lz - 1);
    for (int ch = 0; ch < 4; ++ch) {   // 32-wide n chunks
        __syncthreads();
        if (wn == (ch >> 1)) {
            #pragma unroll
            for (int ni2 = 0; ni2 < 2; ++ni2) {
                int ni = (ch & 1) * 2 + ni2;
                #pragma unroll
                for (int mi = 0; mi < 2; ++mi)
                    #pragma unroll
                    for (int r = 0; r < 8; ++r) {
                        int row  = wm * 32 + mi * 16 + ((lane < 16) ? r : r + 8);
                        int ncol = ni2 * 16 + lane16;
                        sT[row][ncol] = acc[mi][ni][r] + bias[n0 + ch * 32 + ncol];
                    }
            }
        }
        __syncthreads();
        int nl = tid >> 3;
        int l0 = (tid & 7) * 16;
        float* dst = &xt[((size_t)bidx * Dz + (n0 + ch * 32 + nl)) * Lz + lbase + l0];
        #pragma unroll
        for (int i = 0; i < 16; ++i) dst[i] = sT[l0 + i][nl];
    }
}

// ---------------------------------------------------------------------------
// Kernel 2: FFT long-conv. One workgroup per (b,c,d).
// TDM loads x/kernel rows into LDS (tensor_load_to_lds + s_wait_tensorcnt),
// z = x + i*k packed -> 8192-pt complex Stockham FFT in LDS, Hermitian split,
// pointwise product, inverse FFT, + dmat skip term; TDM stores the bf16 y row.
// ---------------------------------------------------------------------------
__device__ float2* wg_fft8192(float2* src, float2* dst, float sign)
{
    int n  = 8192;
    int ls = 0;                 // log2(s)
    while (n > 1) {
        int m = n >> 1;
        __syncthreads();
        float ang0 = sign * 6.283185307179586f / (float)n;
        for (int t = threadIdx.x; t < 4096; t += 256) {
            int s = 1 << ls;
            int p = t >> ls;
            int q = t & (s - 1);
            float2 a = src[q + s * p];
            float2 b = src[q + s * (p + m)];
            float sn, cs;
            __sincosf(ang0 * (float)p, &sn, &cs);
            float2 dif = make_float2(a.x - b.x, a.y - b.y);
            dst[q + s * (2 * p)]     = make_float2(a.x + b.x, a.y + b.y);
            dst[q + s * (2 * p + 1)] = make_float2(dif.x * cs - dif.y * sn,
                                                   dif.x * sn + dif.y * cs);
        }
        float2* tmp = src; src = dst; dst = tmp;
        n = m; ++ls;
    }
    __syncthreads();
    return src;  // buffer holding the result
}

__global__ __launch_bounds__(256) void fftconv_kernel(
    const float* __restrict__ xt, const float* __restrict__ kern,
    const float* __restrict__ dmat, unsigned short* __restrict__ ybf)
{
    extern __shared__ float smem[];
    float2* bufA = (float2*)smem;          // 8192 complex  (64 KB)
    float2* bufB = bufA + 8192;            // 8192 complex  (64 KB)
    float*  xs   = (float*)(bufB + 8192);  // 4096 f32 x row (16 KB)
    float*  ks   = xs + 4096;              // 4096 f32 k row (16 KB)

    const int tid = threadIdx.x;
    const int wid = tid >> 5;
    const int idx = blockIdx.x;
    const int d = idx & (Dz - 1);
    const int c = (idx >> 10) & (Cz - 1);
    const int b = idx >> 12;

    const float* xrow = &xt[((size_t)b * Dz + d) * Lz];
    const float* krow = &kern[((size_t)c * Dz + d) * Lz];

#if USE_TDM_LOAD
    if (wid == 0) {   // one wave drives the Tensor Data Mover
        v4u g0; v8i_t g1;
        v4i_t gz4 = {0, 0, 0, 0};
        v8i_t gz8 = {0, 0, 0, 0, 0, 0, 0, 0};
        tdm_make_desc(xrow, lds_offset_of(xs), Lz, 2u, &g0, &g1);
        __builtin_amdgcn_tensor_load_to_lds(g0, g1, gz4, gz4, gz8, 0);
        tdm_make_desc(krow, lds_offset_of(ks), Lz, 2u, &g0, &g1);
        __builtin_amdgcn_tensor_load_to_lds(g0, g1, gz4, gz4, gz8, 0);
        __builtin_amdgcn_s_wait_tensorcnt(0);
    }
    __syncthreads();
    #pragma unroll 4
    for (int i = 0; i < 16; ++i) {
        int l = tid + i * 256;
        bufA[l]        = make_float2(xs[l], ks[l]);
        bufA[l + 4096] = make_float2(0.f, 0.f);     // zero pad to 2L
    }
#else
    #pragma unroll 4
    for (int i = 0; i < 16; ++i) {
        int l = tid + i * 256;
        float xv = xrow[l];
        bufA[l]        = make_float2(xv, krow[l]);
        bufA[l + 4096] = make_float2(0.f, 0.f);
        xs[l] = xv;
    }
#endif

    float2* Z = wg_fft8192(bufA, bufB, -1.0f);
    float2* Q = (Z == bufA) ? bufB : bufA;

    // Z[j] = Xf[j] + i*Kf[j]; split via Hermitian symmetry, multiply spectra
    for (int t = tid; t < 8192; t += 256) {
        int jn = (8192 - t) & 8191;
        float2 zj = Z[t], zn = Z[jn];
        float2 Xf = make_float2(0.5f * (zj.x + zn.x),  0.5f * (zj.y - zn.y));
        float2 Kf = make_float2(0.5f * (zj.y + zn.y), -0.5f * (zj.x - zn.x));
        Q[t] = make_float2(Xf.x * Kf.x - Xf.y * Kf.y,
                           Xf.x * Kf.y + Xf.y * Kf.x);
    }

    float2* R = wg_fft8192(Q, Z, +1.0f);

    const float dm  = dmat[c * Dz + d];
    const float inv = 1.0f / 8192.0f;
    unsigned short* yrow = &ybf[((size_t)b * CDz + c * Dz + d) * Lz];

#if USE_TDM_STORE
    // stage bf16 row in LDS (reuse ks region; its reads are long done),
    // then DMA it out with the TDM.
    unsigned short* ystage = (unsigned short*)ks;
    #pragma unroll 4
    for (int i = 0; i < 16; ++i) {
        int l = tid + i * 256;
        ystage[l] = f2bf(R[l].x * inv + dm * xs[l]);
    }
    __syncthreads();
    if (wid == 0) {
        v4u g0; v8i_t g1;
        v4i_t gz4 = {0, 0, 0, 0};
        v8i_t gz8 = {0, 0, 0, 0, 0, 0, 0, 0};
        tdm_make_desc(yrow, lds_offset_of(ystage), Lz, 1u, &g0, &g1);
        __builtin_amdgcn_tensor_store_from_lds(g0, g1, gz4, gz4, gz8, 0);
        __builtin_amdgcn_s_wait_tensorcnt(0);
    }
#else
    #pragma unroll 4
    for (int i = 0; i < 16; ++i) {
        int l = tid + i * 256;
        yrow[l] = f2bf(R[l].x * inv + dm * xs[l]);
    }
#endif
}

// ---------------------------------------------------------------------------
// Kernel 3: out_proj  out = y[B,L,CD] @ out_w[CD,D] + out_b
// y is stored K-major ([B, CD, L] bf16): A tiles are loaded contiguous along L
// and transposed through LDS. Same WMMA tile engine as kernel 1.
// ---------------------------------------------------------------------------
__global__ __launch_bounds__(256) void out_proj_wmma(
    const unsigned short* __restrict__ y, const float* __restrict__ w,
    const float* __restrict__ bias, float* __restrict__ out)
{
    __shared__ unsigned short sA[128][34];   // [l][k]
    __shared__ unsigned short sB[128][34];   // [n][k]

    const int tid    = threadIdx.x;
    const int lane   = tid & 31;
    const int wid    = tid >> 5;
    const int wm     = wid & 3;
    const int wn     = wid >> 2;
    const int m0     = blockIdx.y * 128;
    const int n0     = blockIdx.x * 128;
    const int lane16 = lane & 15;
    const int koff   = (lane < 16) ? 0 : 8;
    const int bidx   = m0 >> 12;
    const int lbase  = m0 & (Lz - 1);

    v8f acc[2][4];
    const v8f vzero = {0.f, 0.f, 0.f, 0.f, 0.f, 0.f, 0.f, 0.f};
    for (int mi = 0; mi < 2; ++mi)
        for (int ni = 0; ni < 4; ++ni) acc[mi][ni] = vzero;

    for (int kt = 0; kt < CDz / 32; ++kt) {
        __syncthreads();
        {   // A tile: y[(b*CD + k), lbase + l] bf16, transpose into sA[l][k]
            int flat = tid * 16;
            int kk = flat >> 7, ll = flat & 127;
            const unsigned short* src =
                &y[((size_t)bidx * CDz + kt * 32 + kk) * Lz + lbase + ll];
            #pragma unroll
            for (int i = 0; i < 16; ++i) sA[ll + i][kk] = src[i];
        }
        {   // B tile from out_w[CD, D] -> sB[n][k]
            int flat = tid * 16;
            int kk = flat >> 7, nl = flat & 127;
            const float* src = &w[(size_t)(kt * 32 + kk) * Dz + n0 + nl];
            #pragma unroll
            for (int i = 0; i < 16; ++i) sB[nl + i][kk] = f2bf(src[i]);
        }
        if (kt + 1 < CDz / 32) {
            __builtin_prefetch(&y[((size_t)bidx * CDz + (kt + 1) * 32 + (tid >> 4)) * Lz + lbase], 0, 0);
            __builtin_prefetch(&w[(size_t)((kt + 1) * 32 + (tid >> 4)) * Dz + n0], 0, 0);
        }
        __syncthreads();

        Frag16 a[2], b[4];
        #pragma unroll
        for (int mi = 0; mi < 2; ++mi) {
            int row = wm * 32 + mi * 16 + lane16;
            #pragma unroll
            for (int j = 0; j < 4; ++j) {
                a[mi].w[j]     = *(const unsigned int*)&sA[row][koff + j * 2];
                a[mi].w[j + 4] = *(const unsigned int*)&sA[row][koff + 16 + j * 2];
            }
        }
        #pragma unroll
        for (int ni = 0; ni < 4; ++ni) {
            int col = wn * 64 + ni * 16 + lane16;
            #pragma unroll
            for (int j = 0; j < 4; ++j) {
                b[ni].w[j]     = *(const unsigned int*)&sB[col][koff + j * 2];
                b[ni].w[j + 4] = *(const unsigned int*)&sB[col][koff + 16 + j * 2];
            }
        }
        #pragma unroll
        for (int mi = 0; mi < 2; ++mi)
            #pragma unroll
            for (int ni = 0; ni < 4; ++ni)
                acc[mi][ni] = __builtin_amdgcn_wmma_f32_16x16x32_bf16(
                    false, a[mi].v, false, b[ni].v, (short)0, acc[mi][ni],
                    false, false);
    }

    // epilogue: direct coalesced f32 stores (lanes span contiguous n)
    #pragma unroll
    for (int mi = 0; mi < 2; ++mi)
        #pragma unroll
        for (int ni = 0; ni < 4; ++ni)
            #pragma unroll
            for (int r = 0; r < 8; ++r) {
                int row = m0 + wm * 32 + mi * 16 + ((lane < 16) ? r : r + 8);
                int col = n0 + wn * 64 + ni * 16 + lane16;
                out[(size_t)row * Dz + col] = acc[mi][ni][r] + bias[col];
            }
}

// ---------------------------------------------------------------------------
extern "C" void kernel_launch(void* const* d_in, const int* in_sizes, int n_in,
                              void* d_out, int out_size, void* d_ws, size_t ws_size,
                              hipStream_t stream)
{
    const float* u     = (const float*)d_in[0];
    const float* in_w  = (const float*)d_in[1];
    const float* in_b  = (const float*)d_in[2];
    const float* kern  = (const float*)d_in[3];
    const float* dmat  = (const float*)d_in[4];
    const float* out_w = (const float*)d_in[5];
    const float* out_b = (const float*)d_in[6];
    float* out = (float*)d_out;

    // workspace: xt f32 [B,D,L] (32 MB) + y bf16 [B,CD,L] (64 MB)
    float* xt = (float*)d_ws;
    unsigned short* ybf =
        (unsigned short*)((char*)d_ws + (size_t)Bz * Dz * Lz * sizeof(float));

    dim3 blk(256);
    in_proj_wmma<<<dim3(Dz / 128, Mz / 128), blk, 0, stream>>>(u, in_w, in_b, xt);

    // 2*64KB FFT ping-pong + 16KB x row + 16KB k row = 160 KB dynamic LDS
    size_t fft_lds = 2 * 8192 * sizeof(float2) + 2 * 4096 * sizeof(float);
    fftconv_kernel<<<dim3(Bz * Cz * Dz), blk, fft_lds, stream>>>(xt, kern, dmat, ybf);

    out_proj_wmma<<<dim3(Dz / 128, Mz / 128), blk, 0, stream>>>(ybf, out_w, out_b, out);
}
